// Spatial_AttLayer_2104533975304
// MI455X (gfx1250) — matrix operand order repaired and checked by reading, not compiled, not run on hardware
//
#include <hip/hip_runtime.h>

// ---------------------------------------------------------------------------
// Problem constants (from reference)
// ---------------------------------------------------------------------------
#define NB     8      // batch
#define CIN    256    // feature channels
#define TT     1024   // time steps
#define VV     25     // spatial nodes
#define NH     8      // heads
#define QKD    64     // qk dim
#define VD     64     // v dim
#define CTXT   512    // text channels

// Main-kernel tiling
#define TBLK   8                    // time steps per block
#define NCOL   (TBLK * VV)          // 200 real columns
#define NPAD   208                  // 13 tiles of 16
#define THREADS 512                 // 16 waves (wave32)
#define NWAVES 16

// Workspace layout (bytes)
#define WS_AT  0                            // attn^T bf16 [8][32][32]   16384 B
#define WS_WV  16384                        // Wv bf16 [64][256]         32768 B
#define WS_WO  49152                        // Wo bf16 [256][512]       262144 B

// Dynamic LDS layout. sZT aliases sFT (sFT dead after GEMM1).
#define FPITCH     264                      // padded row pitch of sFT (banks: 528B)
#define SFT_ELEMS  (NPAD * FPITCH)          // [208][264] bf16 (109824 B)
#define SZT_ELEMS  (NPAD * 512)             // [208][512] bf16 (212992 B)
#define SV_OFF     (SZT_ELEMS * 2)          // 212992
#define SV_ELEMS   (VD * TBLK * 32)         // [64][8][32] bf16 (32768 B)
#define SA_OFF     (SV_OFF + SV_ELEMS * 2)  // 245760
#define SA_ELEMS   (NH * 32 * 32)           // [8][32][32] bf16 (16384 B)
#define SMEM_BYTES (SA_OFF + SA_ELEMS * 2)  // 262144 = 256 KB (< 320 KB/WGP)

typedef __attribute__((ext_vector_type(16))) __bf16 bf16x16;
typedef __attribute__((ext_vector_type(8)))  __bf16 bf16x8;
typedef __attribute__((ext_vector_type(8)))  float  f32x8;

// ---------------------------------------------------------------------------
// WMMA + fragment helpers (CDNA5 16x16x32 bf16, fp32 accumulate)
// ---------------------------------------------------------------------------
__device__ __forceinline__ f32x8 wmma_bf16(bf16x16 a, bf16x16 b, f32x8 c) {
  // (neg_a, A, neg_b, B, c_mod, C, reuse_a, reuse_b)
  return __builtin_amdgcn_wmma_f32_16x16x32_bf16(false, a, false, b,
                                                 (short)0, c, false, false);
}

__device__ __forceinline__ bf16x16 combine8(bf16x8 lo, bf16x8 hi) {
  return __builtin_shufflevector(lo, hi, 0, 1, 2, 3, 4, 5, 6, 7,
                                 8, 9, 10, 11, 12, 13, 14, 15);
}

// A fragment (16x32): lane (half=lane>>4, m=lane&15) holds row m,
// K runs 16*half+0..15 contiguous: two adjacent b128 loads.
__device__ __forceinline__ bf16x16 load_frag_a(const __bf16* p) {
  return combine8(*(const bf16x8*)p, *(const bf16x8*)(p + 8));
}
// B fragment (32x16): lane holds column n=lane&15, elems = 16 contiguous K
// starting at 16*half (source stored column-major): two adjacent b128 loads.
__device__ __forceinline__ bf16x16 load_frag_b(const __bf16* p) {
  return combine8(*(const bf16x8*)p, *(const bf16x8*)(p + 8));
}

// ---------------------------------------------------------------------------
// Kernel 0: convert Wv and Wo to bf16 in workspace (L2-resident, reused by
// all 1024 main blocks -> pay the fp32->bf16 cost once, not per fragment).
// ---------------------------------------------------------------------------
__global__ void convert_kernel(const float* __restrict__ Wv,
                               const float* __restrict__ Wo,
                               unsigned short* __restrict__ ws) {
  __bf16* wv = (__bf16*)(ws + WS_WV / 2);
  __bf16* wo = (__bf16*)(ws + WS_WO / 2);
  const int idx = blockIdx.x * blockDim.x + threadIdx.x;
  const int nv = VD * CIN;          // 16384
  const int no = CIN * CTXT;        // 131072
  if (idx < nv) wv[idx] = (__bf16)Wv[idx];
  if (idx < no) wo[idx] = (__bf16)Wo[idx];
}

// ---------------------------------------------------------------------------
// Kernel 1: attention table. Q/K are batch-invariant => compute once.
// One block per head. Writes transposed, zero-padded bf16 aT[h][v'][u] (32x32).
// ---------------------------------------------------------------------------
__global__ void attn_kernel(const float* __restrict__ tf,   // [512][25]
                            const float* __restrict__ Wq, const float* __restrict__ bq,
                            const float* __restrict__ Wk, const float* __restrict__ bk,
                            unsigned short* __restrict__ ws) {
  const int h = blockIdx.x;
  const int tid = threadIdx.x;
  __shared__ float sq[QKD][VV];
  __shared__ float sk[QKD][VV];
  __shared__ float se[VV][VV];

  for (int idx = tid; idx < QKD * VV; idx += 256) {
    const int d = idx / VV, u = idx % VV;
    const int row = h * QKD + d;
    float aq = bq[row], ak = bk[row];
    const float* wqr = Wq + (size_t)row * CTXT;
    const float* wkr = Wk + (size_t)row * CTXT;
    for (int c = 0; c < CTXT; ++c) {
      const float t = tf[c * VV + u];
      aq += wqr[c] * t;
      ak += wkr[c] * t;
    }
    sq[d][u] = aq;
    sk[d][u] = ak;
  }
  __syncthreads();

  for (int idx = tid; idx < VV * VV; idx += 256) {
    const int u = idx / VV, v = idx % VV;
    float e = 0.f;
#pragma unroll 8
    for (int d = 0; d < QKD; ++d) e += sq[d][u] * sk[d][v];
    se[u][v] = e * 0.125f;  // 1/sqrt(64)
  }
  __syncthreads();

  if (tid < VV) {  // softmax over v per row u
    const int u = tid;
    float mx = -3.4e38f;
    for (int v = 0; v < VV; ++v) mx = fmaxf(mx, se[u][v]);
    float s = 0.f;
    for (int v = 0; v < VV; ++v) { float ex = __expf(se[u][v] - mx); se[u][v] = ex; s += ex; }
    const float inv = 1.f / s;
    for (int v = 0; v < VV; ++v) se[u][v] *= inv;
  }
  __syncthreads();

  __bf16* aT = (__bf16*)(ws + WS_AT / 2) + h * 32 * 32;  // [v'][u], zero padded
  for (int idx = tid; idx < 32 * 32; idx += 256) {
    const int v = idx / 32, u = idx % 32;
    const float val = (v < VV && u < VV) ? se[u][v] : 0.f;
    aT[idx] = (__bf16)val;
  }
}

// ---------------------------------------------------------------------------
// Kernel 2: fused  v = Wv@f + bv ; z = attn-mix(v) ; out = Wo@z + bo + f
// Grid: (T/TBLK, N).  512 threads = 16 waves.  All GEMMs via bf16 WMMA.
// ---------------------------------------------------------------------------
__global__ void __launch_bounds__(THREADS)
fused_kernel(const float* __restrict__ feature,
             const float* __restrict__ bv, const float* __restrict__ bo,
             const unsigned short* __restrict__ ws,
             float* __restrict__ out) {
  extern __shared__ char smem[];
  __bf16* sFT = (__bf16*)smem;               // [208 j][264 c] K-contig (dies after GEMM1)
  __bf16* sZT = (__bf16*)smem;               // [208 j][512 zr] K-contig (aliases sFT)
  __bf16* sV  = (__bf16*)(smem + SV_OFF);    // [64 c][8 t][32 u-padded]
  __bf16* sA  = (__bf16*)(smem + SA_OFF);    // [8 h][32 v'][32 u] attn^T padded

  const __bf16* aT  = (const __bf16*)(ws + WS_AT / 2);
  const __bf16* WvB = (const __bf16*)(ws + WS_WV / 2);
  const __bf16* WoB = (const __bf16*)(ws + WS_WO / 2);

  const int t0   = blockIdx.x * TBLK;
  const int n    = blockIdx.y;
  const int tid  = threadIdx.x;
  const int wave = tid >> 5;
  const int lane = tid & 31;
  const int half = (lane >> 4) & 1;
  const int lm   = lane & 15;

  // Async DMA the 16 KB attn table straight into LDS (pure copy, no format
  // conversion -> bypass VGPRs; tracked on ASYNCcnt so it overlaps the heavy
  // fp32->bf16 feature staging below).  LDS dest address = low 32 bits of the
  // flat shared-aperture pointer (ISA 10.2 LDS truncation rule).
  {
    unsigned lds_off = (unsigned)(uintptr_t)sA + (unsigned)tid * 16u;
    const char* gsrc = (const char*)aT + (size_t)tid * 16u;
#pragma unroll
    for (int it = 0; it < (SA_ELEMS * 2) / (THREADS * 16); ++it) {
      asm volatile("global_load_async_to_lds_b128 %0, %1, off"
                   :: "v"(lds_off), "v"(gsrc)
                   : "memory");
      lds_off += THREADS * 16u;
      gsrc    += THREADS * 16u;
    }
  }

  // zero-init sV (u padding 25..31 must be 0)
  for (int i = tid; i < SV_ELEMS; i += THREADS) sV[i] = (__bf16)0.f;

  // Stage feature tile transposed (fp32 -> bf16): sFT[j][c], K contiguous so
  // GEMM1 B-fragments are 2x ds_load_b128.  Global reads stay coalesced
  // (consecutive threads -> consecutive j within a c-row); transposed LDS
  // writes land 528 B apart -> ~2-way bank conflict (row pitch 264).
  const float* fbase = feature + ((size_t)n * CIN * TT + t0) * VV;
  for (int i = tid; i < CIN * NCOL; i += THREADS) {
    const int c = i / NCOL, j = i % NCOL;
    sFT[j * FPITCH + c] = (__bf16)fbase[(size_t)c * (TT * VV) + j];
  }
  asm volatile("s_wait_asynccnt 0x0" ::: "memory");  // attn DMA complete
  __syncthreads();

  // -------- GEMM1: Vmat[64,200] = Wv[64,256] @ F[256,200] + bv ------------
  for (int tile = wave; tile < 4 * (NPAD / 16); tile += NWAVES) {
    const int mt = tile & 3, nt = tile >> 2;   // mt wave-invariant: A hoisted
    f32x8 acc = {};
#pragma unroll
    for (int kt = 0; kt < CIN / 32; ++kt) {
      bf16x16 a = load_frag_a(WvB + (size_t)(mt * 16 + lm) * CIN + kt * 32 + 16 * half);
      bf16x16 b = load_frag_b(sFT + (nt * 16 + lm) * FPITCH + kt * 32 + 16 * half);
      acc = wmma_bf16(a, b, acc);
    }
    const int j = nt * 16 + lm;
    if (j < NCOL) {
      const int t = j / VV, u = j % VV;
#pragma unroll
      for (int r = 0; r < 8; ++r) {
        const int c = mt * 16 + 8 * half + r;
        sV[(c * TBLK + t) * 32 + u] = (__bf16)(acc[r] + bv[c]);
      }
    }
  }
  __syncthreads();

  // -------- Mix: Z[(h,c),t,v'] = sum_u Vmat[c,t,u] * attn[h,u,v'] ---------
  // Per (h,t): [64x25] = [64x25(u pad 32)] @ [25x25 pad 32x32], 1 WMMA/tile.
  // D rows (8 consecutive z-rows at fixed column j) are contiguous in the
  // column-major sZT -> single ds_store_b128 per lane.
  for (int tile = wave; tile < NH * TBLK * 4 * 2; tile += NWAVES) {
    const int h  = tile & 7;
    const int t  = (tile >> 3) & 7;
    const int ct = (tile >> 6) & 3;
    const int vt = (tile >> 8) & 1;
    bf16x16 a = load_frag_a(sV + ((ct * 16 + lm) * TBLK + t) * 32 + 16 * half);
    bf16x16 b = load_frag_b(sA + (h * 32 + vt * 16 + lm) * 32 + 16 * half);
    f32x8 acc = {};
    acc = wmma_bf16(a, b, acc);
    const int vp = vt * 16 + lm;
    if (vp < VV) {
      bf16x8 zv;
#pragma unroll
      for (int r = 0; r < 8; ++r) zv[r] = (__bf16)acc[r];
      *(bf16x8*)(sZT + (size_t)(t * VV + vp) * 512 + h * VD + ct * 16 + 8 * half) = zv;
    }
  }
  __syncthreads();

  // -------- GEMM3: out[256,200] = Wo[256,512] @ Z[512,200] + bo + residual
  for (int tile = wave; tile < 16 * (NPAD / 16); tile += NWAVES) {
    const int ot = tile & 15, nt = tile >> 4;  // ot wave-invariant: A hoisted
    f32x8 acc = {};
#pragma unroll
    for (int kt = 0; kt < CTXT / 32; ++kt) {
      bf16x16 a = load_frag_a(WoB + (size_t)(ot * 16 + lm) * CTXT + kt * 32 + 16 * half);
      bf16x16 b = load_frag_b(sZT + (size_t)(nt * 16 + lm) * 512 + kt * 32 + 16 * half);
      acc = wmma_bf16(a, b, acc);
    }
    const int j = nt * 16 + lm;
    if (j < NCOL) {
      const int t = t0 + j / VV, v = j % VV;
#pragma unroll
      for (int r = 0; r < 8; ++r) {
        const int o = ot * 16 + 8 * half + r;
        const size_t g = (((size_t)n * CIN + o) * TT + t) * VV + v;
        out[g] = acc[r] + bo[o] + feature[g];
      }
    }
  }
}

// ---------------------------------------------------------------------------
// Host launcher
// ---------------------------------------------------------------------------
extern "C" void kernel_launch(void* const* d_in, const int* in_sizes, int n_in,
                              void* d_out, int out_size, void* d_ws, size_t ws_size,
                              hipStream_t stream) {
  (void)in_sizes; (void)n_in; (void)out_size; (void)ws_size;
  const float* feature = (const float*)d_in[0];
  const float* tf      = (const float*)d_in[1];
  const float* Wq      = (const float*)d_in[2];
  const float* bq      = (const float*)d_in[3];
  const float* Wk      = (const float*)d_in[4];
  const float* bk      = (const float*)d_in[5];
  const float* Wv      = (const float*)d_in[6];
  const float* bv      = (const float*)d_in[7];
  const float* Wo      = (const float*)d_in[8];
  const float* bo      = (const float*)d_in[9];
  float* out = (float*)d_out;
  unsigned short* ws = (unsigned short*)d_ws;  // ~304 KB: attn^T + bf16 weights

  convert_kernel<<<(CIN * CTXT + 255) / 256, 256, 0, stream>>>(Wv, Wo, ws);
  attn_kernel<<<NH, 256, 0, stream>>>(tf, Wq, bq, Wk, bk, ws);
  fused_kernel<<<dim3(TT / TBLK, NB), THREADS, SMEM_BYTES, stream>>>(
      feature, bv, bo, ws, out);
}
